// Net_37323265802379
// MI455X (gfx1250) — matrix operand order repaired and checked by reading, not compiled, not run on hardware
//
#include <hip/hip_runtime.h>
#include <hip/hip_bf16.h>

typedef __attribute__((ext_vector_type(16))) _Float16 v16h;
typedef __attribute__((ext_vector_type(8)))  float    v8f;
typedef int v4i_vs __attribute__((vector_size(16)));   // matches async-builtin param type

#define LOG2F_C 0.69314718055994530942f
#define PI_F    3.14159265358979323846f

#if defined(__has_builtin)
#  if __has_builtin(__builtin_amdgcn_global_load_async_to_lds_b128)
#    define HAS_ASYNC 1
#  endif
#endif
#ifndef HAS_ASYNC
#  define HAS_ASYNC 0
#endif

// shifted softplus: softplus(x) - ln2, numerically stable
__device__ __forceinline__ float ssp(float x) {
    return fmaxf(x, 0.0f) + logf(1.0f + expf(-fabsf(x))) - LOG2F_C;
}

// ---- WMMA fragment loaders (16x16x32 f16, wave32, contiguous k = 16*(lane/16) + s in A and B) ----
// A: 16xK row-major in LDS, lda elements per row.
__device__ __forceinline__ v16h load_frag_a(const _Float16* A, int lda, int lane, int k0) {
    int m  = lane & 15;
    int kb = k0 + ((lane >> 4) << 4);
    const _Float16* p = A + m * lda + kb;
    v16h a;
#pragma unroll
    for (int e = 0; e < 8; ++e) a[e] = p[e];
#pragma unroll
    for (int e = 0; e < 8; ++e) a[8 + e] = p[8 + e];
    return a;
}

// B pre-swizzled to fragment order:
//   swz[((k>>5)*N + n)*32 + ((k>>4)&1)*16 + (k&15)] = B[k][n]
// lane reads 32 contiguous bytes -> 2x ds_load_b128.
__device__ __forceinline__ v16h load_frag_b_swz(const _Float16* B, int N, int lane,
                                                int k0, int n0) {
    int n = n0 + (lane & 15);
    const _Float16* p = B + (((k0 >> 5) * N + n) * 2 + (lane >> 4)) * 16;
    v16h b;
#pragma unroll
    for (int e = 0; e < 16; ++e) b[e] = p[e];
    return b;
}

__device__ __forceinline__ v8f wmma_f16(v16h a, v16h b, v8f c) {
    return __builtin_amdgcn_wmma_f32_16x16x32_f16(false, a, false, b, (short)0, c, false, false);
}

__device__ __forceinline__ void atomic_add_f32(float* p, float v) {
    __hip_atomic_fetch_add(p, v, __ATOMIC_RELAXED, __HIP_MEMORY_SCOPE_AGENT);
}

// pack two f32 -> two f16 -> one 32-bit LDS store (dst must be 4B aligned)
__device__ __forceinline__ void store2_f16(_Float16* dst, float a, float b) {
    union { _Float16 h[2]; unsigned u; } t;
    t.h[0] = (_Float16)a; t.h[1] = (_Float16)b;
    *reinterpret_cast<unsigned*>(dst) = t.u;
}

// ---- LDS staging of pre-swizzled f16 data: async copy (CDNA5) or uint4 fallback ----
__device__ __forceinline__ void stage_lds_f16(const _Float16* __restrict__ g,
                                              _Float16* lds, int nelem, int tid) {
#if HAS_ASYNC
    for (int off = tid * 8; off < nelem; off += 256 * 8)
        __builtin_amdgcn_global_load_async_to_lds_b128(
            (v4i_vs*)(g + off), (v4i_vs*)(lds + off), 0, 0);
#else
    const uint4* src = reinterpret_cast<const uint4*>(g);
    uint4* dst = reinterpret_cast<uint4*>(lds);
    for (int i = tid; i < nelem / 8; i += 256) dst[i] = src[i];
#endif
}

__device__ __forceinline__ void wait_async() {
#if HAS_ASYNC
#  if defined(__has_builtin) && __has_builtin(__builtin_amdgcn_s_wait_asynccnt)
    __builtin_amdgcn_s_wait_asynccnt(0);
#  else
    asm volatile("s_wait_asynccnt 0" ::: "memory");
#  endif
#endif
}

// stage 16x128 f32 activations -> f16 LDS tile (paired 32-bit stores)
__device__ __forceinline__ void stage_a_tile(const float* __restrict__ src,
                                             _Float16* dst, int tid) {
    for (int i = tid * 2; i < 16 * 128; i += 512)
        store2_f16(&dst[i], src[i], src[i + 1]);
}

// ============================ setup kernels ============================

// convert KxN f32 row-major weights to swizzled f16 (B-fragment order) in global ws
__global__ void __launch_bounds__(256) k_prep_w(const float* __restrict__ W,
                                                _Float16* __restrict__ dst,
                                                int K, int N, int krows) {
    int base = (blockIdx.x * 256 + threadIdx.x) * 8;
    if (base >= K * N) return;
    int e0 = base & 15;            // 0 or 8
    int kh = (base >> 4) & 1;
    int n  = (base >> 5) % N;
    int kt = base / (32 * N);
    int kbase = kt * 32 + kh * 16 + e0;
    union { _Float16 h[8]; uint4 v; } tmp;
#pragma unroll
    for (int j = 0; j < 8; ++j) {
        int k = kbase + j;
        tmp.h[j] = (k < krows) ? (_Float16)W[k * N + n] : (_Float16)0.0f;
    }
    *reinterpret_cast<uint4*>(dst + base) = tmp.v;
}

__global__ void __launch_bounds__(256) k_init_r(const float* __restrict__ emb,
                                                const int* __restrict__ z,
                                                float* __restrict__ r, int total) {
    int i = blockIdx.x * 256 + threadIdx.x;
    if (i >= total) return;
    int atom = i >> 7, c = i & 127;
    r[i] = emb[z[atom] * 128 + c];
}

__global__ void __launch_bounds__(256) k_zero(float* __restrict__ p, int n) {
    int i = blockIdx.x * 256 + threadIdx.x;
    if (i < n) p[i] = 0.0f;
}

// per-edge: distance -> gaussian basis (f16, padded 50->64) + cosine cutoff
__global__ void __launch_bounds__(256) k_edge_geom(const float* __restrict__ xyz,
                                                   const int* __restrict__ a,
                                                   _Float16* __restrict__ gq,
                                                   float* __restrict__ Cc, int E) {
    int e = blockIdx.x * 256 + threadIdx.x;
    if (e >= E) return;
    int a0 = a[2 * e], a1 = a[2 * e + 1];
    float dx = xyz[3 * a0 + 0] - xyz[3 * a1 + 0];
    float dy = xyz[3 * a0 + 1] - xyz[3 * a1 + 1];
    float dz = xyz[3 * a0 + 2] - xyz[3 * a1 + 2];
    float d = sqrtf(dx * dx + dy * dy + dz * dz);
    Cc[e] = 0.5f * (cosf(d * (PI_F / 5.0f)) + 1.0f);
    const float step = 5.0f / 49.0f;
    const float coef = -0.5f / (step * step);
    union { _Float16 h[64]; uint4 v[8]; } buf;
#pragma unroll 1
    for (int j = 0; j < 50; ++j) {
        float t = d - step * (float)j;
        buf.h[j] = (_Float16)expf(coef * t * t);
    }
#pragma unroll
    for (int j = 50; j < 64; ++j) buf.h[j] = (_Float16)0.0f;
    uint4* gp = reinterpret_cast<uint4*>(gq + (size_t)e * 64);
#pragma unroll
    for (int v = 0; v < 8; ++v) gp[v] = buf.v[v];
}

// ============================ h = r @ win ; zero y ============================

__global__ void __launch_bounds__(256) k_in2f(const float* __restrict__ r,
                                              const _Float16* __restrict__ winS,
                                              float* __restrict__ h,
                                              float* __restrict__ y) {
    __shared__ alignas(16) _Float16 wbuf[128 * 128];
    __shared__ alignas(16) _Float16 atile[16 * 128];
    const int tid = threadIdx.x, lane = tid & 31, wv = tid >> 5;
    const int i0 = blockIdx.x * 16;

    stage_lds_f16(winS, wbuf, 128 * 128, tid);
    stage_a_tile(r + (size_t)i0 * 128, atile, tid);
    {   // zero y rows (float4)
        float4* yp = reinterpret_cast<float4*>(y + (size_t)i0 * 128);
        float4 zv = make_float4(0.f, 0.f, 0.f, 0.f);
        for (int i = tid; i < 16 * 128 / 4; i += 256) yp[i] = zv;
    }
    wait_async();
    __syncthreads();

    v8f acc = {};
#pragma unroll
    for (int k0 = 0; k0 < 128; k0 += 32) {
        v16h af = load_frag_a(atile, 128, lane, k0);
        v16h bf = load_frag_b_swz(wbuf, 128, lane, k0, wv * 16);
        acc = wmma_f16(af, bf, acc);
    }
    const int n = wv * 16 + (lane & 15);
    const int mb = (lane >> 4) << 3;
#pragma unroll
    for (int i = 0; i < 8; ++i)
        h[(size_t)(i0 + mb + i) * 128 + n] = acc[i];
}

// ===== fused edge filter: W = ssp(g@fw1+b1)@fw2+b2, W*=C, scatter into y =====

__global__ void __launch_bounds__(256) k_edge_filter(const _Float16* __restrict__ gq,
                                                     const float* __restrict__ Cc,
                                                     const _Float16* __restrict__ fw1S,
                                                     const float* __restrict__ fb1,
                                                     const _Float16* __restrict__ fw2S,
                                                     const float* __restrict__ fb2,
                                                     const int* __restrict__ a,
                                                     const float* __restrict__ h,
                                                     float* __restrict__ y) {
    __shared__ alignas(16) _Float16 wbuf1[64 * 128];    // fw1 swz (padded 50->64)
    __shared__ alignas(16) _Float16 wbuf2[128 * 128];   // fw2 swz
    __shared__ alignas(16) _Float16 gtile[16 * 64];
    __shared__ alignas(16) _Float16 h1[16 * 128];
    __shared__ alignas(16) float    Wt[16 * 128];
    __shared__ float    Ct[16];
    const int tid = threadIdx.x, lane = tid & 31, wv = tid >> 5;
    const int e0 = blockIdx.x * 16;

    stage_lds_f16(fw1S, wbuf1, 64 * 128, tid);
    stage_lds_f16(fw2S, wbuf2, 128 * 128, tid);
    stage_lds_f16(gq + (size_t)e0 * 64, gtile, 16 * 64, tid);
    if (tid < 16) Ct[tid] = Cc[e0 + tid];
    wait_async();
    __syncthreads();

    const int n  = wv * 16 + (lane & 15);
    const int mb = (lane >> 4) << 3;

    // GEMM1: [16,64] @ [64,128]
    v8f acc = {};
#pragma unroll
    for (int k0 = 0; k0 < 64; k0 += 32) {
        v16h af = load_frag_a(gtile, 64, lane, k0);
        v16h bf = load_frag_b_swz(wbuf1, 128, lane, k0, wv * 16);
        acc = wmma_f16(af, bf, acc);
    }
    {
        float b1 = fb1[n];
#pragma unroll
        for (int i = 0; i < 8; ++i)
            h1[(mb + i) * 128 + n] = (_Float16)ssp(acc[i] + b1);
    }
    __syncthreads();

    // GEMM2: [16,128] @ [128,128], then * cutoff
    acc = (v8f){};
#pragma unroll
    for (int k0 = 0; k0 < 128; k0 += 32) {
        v16h af = load_frag_a(h1, 128, lane, k0);
        v16h bf = load_frag_b_swz(wbuf2, 128, lane, k0, wv * 16);
        acc = wmma_f16(af, bf, acc);
    }
    {
        float b2 = fb2[n];
#pragma unroll
        for (int i = 0; i < 8; ++i)
            Wt[(mb + i) * 128 + n] = (acc[i] + b2) * Ct[mb + i];
    }
    __syncthreads();

    // symmetric gather-filter-scatter: 16 threads per edge, 8 cols per thread
    const int row = tid >> 4, cg = tid & 15;
    const int e  = e0 + row;
    const int i0a = a[2 * e], i1a = a[2 * e + 1];
    const float* h0 = h + (size_t)i0a * 128;
    const float* hp = h + (size_t)i1a * 128;
    float* y0 = y + (size_t)i0a * 128;
    float* y1 = y + (size_t)i1a * 128;
#pragma unroll
    for (int j = 0; j < 8; ++j) {
        int c = cg * 8 + j;
        float w = Wt[row * 128 + c];
        atomic_add_f32(&y0[c], hp[c] * w);
        atomic_add_f32(&y1[c], h0[c] * w);
    }
}

// ===== atom update: r += ssp(y@wout1+b)@wout2+b =====

__global__ void __launch_bounds__(256) k_atom_update(const float* __restrict__ y,
                                                     const _Float16* __restrict__ wout1S,
                                                     const float* __restrict__ bout1,
                                                     const _Float16* __restrict__ wout2S,
                                                     const float* __restrict__ bout2,
                                                     float* __restrict__ r) {
    __shared__ alignas(16) _Float16 wbuf[128 * 128];   // wout1 then wout2 (swz)
    __shared__ alignas(16) _Float16 atile[16 * 128];
    __shared__ alignas(16) _Float16 h1[16 * 128];
    const int tid = threadIdx.x, lane = tid & 31, wv = tid >> 5;
    const int i0 = blockIdx.x * 16;

    stage_lds_f16(wout1S, wbuf, 128 * 128, tid);
    stage_a_tile(y + (size_t)i0 * 128, atile, tid);
    wait_async();
    __syncthreads();

    const int n  = wv * 16 + (lane & 15);
    const int mb = (lane >> 4) << 3;

    v8f acc = {};
#pragma unroll
    for (int k0 = 0; k0 < 128; k0 += 32) {
        v16h af = load_frag_a(atile, 128, lane, k0);
        v16h bf = load_frag_b_swz(wbuf, 128, lane, k0, wv * 16);
        acc = wmma_f16(af, bf, acc);
    }
    {
        float b1 = bout1[n];
#pragma unroll
        for (int i = 0; i < 8; ++i)
            h1[(mb + i) * 128 + n] = (_Float16)ssp(acc[i] + b1);
    }
    __syncthreads();

    stage_lds_f16(wout2S, wbuf, 128 * 128, tid);
    wait_async();
    __syncthreads();

    acc = (v8f){};
#pragma unroll
    for (int k0 = 0; k0 < 128; k0 += 32) {
        v16h af = load_frag_a(h1, 128, lane, k0);
        v16h bf = load_frag_b_swz(wbuf, 128, lane, k0, wv * 16);
        acc = wmma_f16(af, bf, acc);
    }
    {
        float b2 = bout2[n];
#pragma unroll
        for (int i = 0; i < 8; ++i) {
            size_t idx = (size_t)(i0 + mb + i) * 128 + n;
            r[idx] = r[idx] + acc[i] + b2;
        }
    }
}

// ===== head: out[mol] += ssp(r@wa1+ba1)@wa2 + ba2 =====

__global__ void __launch_bounds__(256) k_head(const float* __restrict__ r,
                                              const _Float16* __restrict__ wa1S,
                                              const float* __restrict__ ba1,
                                              const float* __restrict__ wa2,
                                              const float* __restrict__ ba2,
                                              const int* __restrict__ mol,
                                              float* __restrict__ out) {
    __shared__ alignas(16) _Float16 wbuf[128 * 64];
    __shared__ alignas(16) _Float16 atile[16 * 128];
    __shared__ alignas(16) float    h2[16 * 64];
    __shared__ float    red[256];
    const int tid = threadIdx.x, lane = tid & 31, wv = tid >> 5;
    const int i0 = blockIdx.x * 16;

    stage_lds_f16(wa1S, wbuf, 128 * 64, tid);
    stage_a_tile(r + (size_t)i0 * 128, atile, tid);
    wait_async();
    __syncthreads();

    if (wv < 4) {  // 64 output cols -> 4 N-tiles; wave-uniform branch, EXEC all-1s inside
        v8f acc = {};
#pragma unroll
        for (int k0 = 0; k0 < 128; k0 += 32) {
            v16h af = load_frag_a(atile, 128, lane, k0);
            v16h bf = load_frag_b_swz(wbuf, 64, lane, k0, wv * 16);
            acc = wmma_f16(af, bf, acc);
        }
        const int n  = wv * 16 + (lane & 15);
        const int mb = (lane >> 4) << 3;
        float b1 = ba1[n];
#pragma unroll
        for (int i = 0; i < 8; ++i)
            h2[(mb + i) * 64 + n] = ssp(acc[i] + b1);
    }
    __syncthreads();

    // second layer: length-64 dot with wa2, 16 threads per atom
    const int row = tid >> 4, sg = tid & 15;
    float p = 0.0f;
#pragma unroll
    for (int j = 0; j < 4; ++j) {
        int c = sg * 4 + j;
        p += h2[row * 64 + c] * wa2[c];
    }
    red[tid] = p;
    __syncthreads();

    if (tid < 16) {
        float s = ba2[0];
#pragma unroll
        for (int k = 0; k < 16; ++k) s += red[tid * 16 + k];
        atomic_add_f32(&out[mol[i0 + tid]], s);
    }
}

// ============================ launcher ============================

extern "C" void kernel_launch(void* const* d_in, const int* in_sizes, int n_in,
                              void* d_out, int out_size, void* d_ws, size_t ws_size,
                              hipStream_t stream) {
    const float* xyz   = (const float*)d_in[0];
    const float* emb   = (const float*)d_in[1];
    const float* fw1   = (const float*)d_in[2];
    const float* fb1   = (const float*)d_in[3];
    const float* fw2   = (const float*)d_in[4];
    const float* fb2   = (const float*)d_in[5];
    const float* win   = (const float*)d_in[6];
    const float* wout1 = (const float*)d_in[7];
    const float* bout1 = (const float*)d_in[8];
    const float* wout2 = (const float*)d_in[9];
    const float* bout2 = (const float*)d_in[10];
    const float* wa1   = (const float*)d_in[11];
    const float* ba1   = (const float*)d_in[12];
    const float* wa2   = (const float*)d_in[13];
    const float* ba2   = (const float*)d_in[14];
    const int*   z     = (const int*)d_in[15];
    const int*   a     = (const int*)d_in[16];
    const int*   mol   = (const int*)d_in[17];

    const int N = in_sizes[15];          // 131072 atoms
    const int E = in_sizes[16] / 2;      // 524288 edges
    const int T = 3;
    const int M = 128 * 128;             // 16384

    // workspace layout
    float*     r  = (float*)d_ws;                         // N*128 f32
    float*     h  = r + (size_t)N * 128;                  // N*128 f32
    float*     y  = h + (size_t)N * 128;                  // N*128 f32
    _Float16*  gq = (_Float16*)(y + (size_t)N * 128);     // E*64 f16
    float*     Cc = (float*)(gq + (size_t)E * 64);        // E f32
    // pre-swizzled f16 weights
    _Float16* winS   = (_Float16*)(Cc + E);               // 3*16384
    _Float16* fw1S   = winS + 3 * M;                      // 3*8192
    _Float16* fw2S   = fw1S + 3 * (M / 2);                // 3*16384
    _Float16* wout1S = fw2S + 3 * M;                      // 3*16384
    _Float16* wout2S = wout1S + 3 * M;                    // 3*16384
    _Float16* wa1S   = wout2S + 3 * M;                    // 8192
    float*    out    = (float*)d_out;

    // one-time weight prep (runs every call; deterministic)
    for (int t = 0; t < T; ++t) {
        k_prep_w<<<8, 256, 0, stream>>>(win   + (size_t)t * M,      winS   + (size_t)t * M,      128, 128, 128);
        k_prep_w<<<4, 256, 0, stream>>>(fw1   + (size_t)t * 50*128, fw1S   + (size_t)t * (M/2),  64,  128, 50);
        k_prep_w<<<8, 256, 0, stream>>>(fw2   + (size_t)t * M,      fw2S   + (size_t)t * M,      128, 128, 128);
        k_prep_w<<<8, 256, 0, stream>>>(wout1 + (size_t)t * M,      wout1S + (size_t)t * M,      128, 128, 128);
        k_prep_w<<<8, 256, 0, stream>>>(wout2 + (size_t)t * M,      wout2S + (size_t)t * M,      128, 128, 128);
    }
    k_prep_w<<<4, 256, 0, stream>>>(wa1, wa1S, 128, 64, 128);

    k_init_r<<<(N * 128 + 255) / 256, 256, 0, stream>>>(emb, z, r, N * 128);
    k_zero<<<(out_size + 255) / 256, 256, 0, stream>>>(out, out_size);
    k_edge_geom<<<(E + 255) / 256, 256, 0, stream>>>(xyz, a, gq, Cc, E);

    for (int t = 0; t < T; ++t) {
        k_in2f<<<N / 16, 256, 0, stream>>>(r, winS + (size_t)t * M, h, y);
        k_edge_filter<<<E / 16, 256, 0, stream>>>(gq, Cc,
                                                  fw1S + (size_t)t * (M / 2),
                                                  fb1 + (size_t)t * 128,
                                                  fw2S + (size_t)t * M,
                                                  fb2 + (size_t)t * 128,
                                                  a, h, y);
        k_atom_update<<<N / 16, 256, 0, stream>>>(y,
                                                  wout1S + (size_t)t * M,
                                                  bout1 + (size_t)t * 128,
                                                  wout2S + (size_t)t * M,
                                                  bout2 + (size_t)t * 128,
                                                  r);
    }

    k_head<<<N / 16, 256, 0, stream>>>(r, wa1S, ba1, wa2, ba2, mol, out);
}
